// GraphDecoder_7464653161134
// MI455X (gfx1250) — matrix operand order repaired and checked by reading, not compiled, not run on hardware
//
#include <hip/hip_runtime.h>
#include <hip/hip_bf16.h>

typedef __attribute__((ext_vector_type(16))) __bf16 v16bf;
typedef __attribute__((ext_vector_type(8)))  float  v8f;

#define Dk   128
#define Hh   4
#define HOk  512
#define Rr   3
#define Ak   64
#define Lk   32
#define Bk   8
#define Ck   1024
#define Nk   (Bk*Ck)          // 8192
#define Ek   (Nk + Nk*Lk)     // 270336
#define EPSf 1e-5f

// ---------------- CSR build (edges are round-invariant) ----------------
__global__ void k_deg_init(int* __restrict__ deg) {
    int i = blockIdx.x * 256 + threadIdx.x;
    if (i < Nk) deg[i] = 1;                       // self edge
}

__global__ void k_count(const int* __restrict__ ids, int* __restrict__ deg) {
    int t = blockIdx.x * 256 + threadIdx.x;
    if (t >= Nk * Lk) return;
    int n   = t / Lk;
    int dst = (n / Ck) * Ck + ids[t];
    atomicAdd(&deg[dst], 1);
}

__global__ __launch_bounds__(256)
void k_scan(const int* __restrict__ deg, int* __restrict__ offs, int* __restrict__ cur) {
    __shared__ int tot[256];
    const int tid = threadIdx.x;
    const int PER = Nk / 256;                     // 32
    int base = tid * PER, s = 0;
    for (int i = 0; i < PER; ++i) s += deg[base + i];
    tot[tid] = s;
    __syncthreads();
    if (tid == 0) {
        int run = 0;
        for (int j = 0; j < 256; ++j) { int t = tot[j]; tot[j] = run; run += t; }
        offs[Nk] = run;
    }
    __syncthreads();
    int off = tot[tid];
    for (int i = 0; i < PER; ++i) {
        offs[base + i] = off; cur[base + i] = off; off += deg[base + i];
    }
}

__global__ void k_fill(const int* __restrict__ ids, const float* __restrict__ rel,
                       int* __restrict__ cur, int* __restrict__ csrc,
                       float* __restrict__ crx, float* __restrict__ cry) {
    int t = blockIdx.x * 256 + threadIdx.x;
    if (t >= Nk + Nk * Lk) return;
    if (t < Nk) {                                 // self edge, edge_attr = 0
        int pos = atomicAdd(&cur[t], 1);
        csrc[pos] = t; crx[pos] = 0.f; cry[pos] = 0.f;
    } else {
        int u = t - Nk;
        int n = u / Lk, l = u - n * Lk;
        int dst = (n / Ck) * Ck + ids[u];
        int pos = atomicAdd(&cur[dst], 1);
        csrc[pos] = n;
        crx[pos] = rel[n * (2 * Lk) + 2 * l];
        cry[pos] = rel[n * (2 * Lk) + 2 * l + 1];
    }
}

// -------- rank-2 edge matrix: M[2][HO] = We_emb[r] @ Wedge[r] ----------
__global__ void k_edgeM(const float* __restrict__ Wemb, const float* __restrict__ Wedge,
                        float* __restrict__ M) {
    int idx = blockIdx.x * blockDim.x + threadIdx.x;
    if (idx >= 2 * HOk) return;
    int j = idx >> 9, dim = idx & (HOk - 1);
    float s = 0.f;
    for (int d = 0; d < Dk; ++d) s += Wemb[j * Dk + d] * Wedge[d * HOk + dim];
    M[idx] = s;
}

// ------------------- WMMA bf16 GEMM: Y = X@W (+Res) --------------------
// grid = (NC/64, Nrows/16), block = 128 (4 waves, 16 N-cols each)
template<int K, int NC, bool RES>
__global__ __launch_bounds__(128)
void k_gemm(const float* __restrict__ X, const float* __restrict__ W,
            const float* __restrict__ Res, float* __restrict__ Y) {
    constexpr int KP = K + 4;                     // pad: conflict-free column reads
    __shared__ float sX[16 * KP];
    const int tid  = threadIdx.x;
    const int lane = tid & 31;
    const int wave = tid >> 5;
    const int tm   = blockIdx.y * 16;
    const int tn   = blockIdx.x * 64 + wave * 16;

    constexpr int K4 = K / 4;                     // float4-vectorized A stage
    for (int idx = tid; idx < 16 * K4; idx += 128) {
        int r = idx / K4, c = idx - r * K4;
        float4 t = reinterpret_cast<const float4*>(X + (tm + r) * K)[c];
        *reinterpret_cast<float4*>(&sX[r * KP + 4 * c]) = t;
    }
    __syncthreads();

    const int m  = lane & 15;
    const int hi = lane >> 4;                     // lane half selects K sub-block
    v8f acc = {};
    #pragma unroll 4
    for (int kb = 0; kb < K; kb += 32) {
        v16bf a, b;
        #pragma unroll
        for (int v = 0; v < 8; ++v) {
            // A 16x32 bf16 (ISA 7.12.2): half0 K{0-7,16-23}, half1 K{8-15,24-31}
            int ka = kb + ((v < 4) ? (2 * v) : (16 + 2 * (v - 4))) + (hi ? 8 : 0);
            a[2 * v]     = (__bf16)sX[m * KP + ka];
            a[2 * v + 1] = (__bf16)sX[m * KP + ka + 1];
            // B 32x16 bf16: half0 K{0-7,16-23}, half1 K{8-15,24-31}, 2 K per VGPR
            int kw = kb + ((v < 4) ? (2 * v) : (16 + 2 * (v - 4))) + (hi ? 8 : 0);
            b[2 * v]     = (__bf16)W[kw * NC + tn + m];
            b[2 * v + 1] = (__bf16)W[(kw + 1) * NC + tn + m];
        }
        acc = __builtin_amdgcn_wmma_f32_16x16x32_bf16(false, a, false, b,
                                                      (short)0, acc, false, false);
    }
    #pragma unroll
    for (int r = 0; r < 8; ++r) {                 // C layout: VGPR r -> M=r / M=8+r
        int row = tm + r + (hi ? 8 : 0);
        int col = tn + m;
        float o = acc[r];
        if constexpr (RES) o += Res[row * NC + col];
        Y[row * NC + col] = o;
    }
}

// ------ per-dst online-softmax attention (1 wave32 per destination) ----
// Lane l owns dims [16l, 16l+16) -> 4x b128 loads/row; head = l>>3.
__global__ __launch_bounds__(256)
void k_attn(const float* __restrict__ q, const float* __restrict__ kk,
            const float* __restrict__ vv, const float* __restrict__ skip,
            const float* __restrict__ M, const int* __restrict__ offs,
            const int* __restrict__ csrc, const float* __restrict__ crx,
            const float* __restrict__ cry, float* __restrict__ attn) {
    const int wave = threadIdx.x >> 5, lane = threadIdx.x & 31;
    const int dst = blockIdx.x * 8 + wave;
    if (dst >= Nk) return;
    const int dbase = lane * 16;

    float4 qv[4], m0v[4], m1v[4], acc[4];
    const float4* qp  = reinterpret_cast<const float4*>(q + dst * HOk + dbase);
    const float4* m0p = reinterpret_cast<const float4*>(M + dbase);
    const float4* m1p = reinterpret_cast<const float4*>(M + HOk + dbase);
    #pragma unroll
    for (int i = 0; i < 4; ++i) {
        qv[i] = qp[i]; m0v[i] = m0p[i]; m1v[i] = m1p[i];
        acc[i] = make_float4(0.f, 0.f, 0.f, 0.f);
    }
    float mh = -3.0e38f, sh = 0.f;
    const float scale = 0.08838834764831845f;     // 1/sqrt(128)
    const int s = offs[dst], e = offs[dst + 1];

    for (int ep = s; ep < e; ++ep) {
        int   src = csrc[ep];
        float rx = crx[ep], ry = cry[ep];
        if (ep + 1 < e) {                         // hide L2 latency on the gather
            int ns = csrc[ep + 1];
            __builtin_prefetch(kk + ns * HOk + dbase, 0, 0);
            __builtin_prefetch(vv + ns * HOk + dbase, 0, 0);
        }
        const float4* kp = reinterpret_cast<const float4*>(kk + src * HOk + dbase);
        float4 er[4];
        float p = 0.f;
        #pragma unroll
        for (int i = 0; i < 4; ++i) {             // alpha partial: q . (k[src]+e)
            float4 kv = kp[i];
            float4 ee;
            ee.x = rx * m0v[i].x + ry * m1v[i].x;
            ee.y = rx * m0v[i].y + ry * m1v[i].y;
            ee.z = rx * m0v[i].z + ry * m1v[i].z;
            ee.w = rx * m0v[i].w + ry * m1v[i].w;
            er[i] = ee;
            p += qv[i].x * (kv.x + ee.x) + qv[i].y * (kv.y + ee.y)
               + qv[i].z * (kv.z + ee.z) + qv[i].w * (kv.w + ee.w);
        }
        p += __shfl_xor(p, 1);                    // reduce within 8-lane head group
        p += __shfl_xor(p, 2);
        p += __shfl_xor(p, 4);
        float al = p * scale;
        float nm = fmaxf(mh, al);
        float so = __expf(mh - nm);
        float w  = __expf(al - nm);
        sh = sh * so + w;
        mh = nm;
        const float4* vp = reinterpret_cast<const float4*>(vv + src * HOk + dbase);
        #pragma unroll
        for (int i = 0; i < 4; ++i) {             // acc rescale + weighted (v[src]+e)
            float4 vl = vp[i];
            acc[i].x = acc[i].x * so + w * (vl.x + er[i].x);
            acc[i].y = acc[i].y * so + w * (vl.y + er[i].y);
            acc[i].z = acc[i].z * so + w * (vl.z + er[i].z);
            acc[i].w = acc[i].w * so + w * (vl.w + er[i].w);
        }
    }
    const float4* sp = reinterpret_cast<const float4*>(skip + dst * HOk + dbase);
    float4* op = reinterpret_cast<float4*>(attn + dst * HOk + dbase);
    float inv = 1.f / sh;
    #pragma unroll
    for (int i = 0; i < 4; ++i) {                 // normalize + root skip
        float4 sk = sp[i], o;
        o.x = acc[i].x * inv + sk.x;
        o.y = acc[i].y * inv + sk.y;
        o.z = acc[i].z * inv + sk.z;
        o.w = acc[i].w * inv + sk.w;
        op[i] = o;
    }
}

// --------------------------- RMSNorm (wave/row) ------------------------
template<int W_>
__global__ __launch_bounds__(256)
void k_rmsnorm(const float* __restrict__ X, const float* __restrict__ w,
               float* __restrict__ Y) {
    const int wave = threadIdx.x >> 5, lane = threadIdx.x & 31;
    const int row = blockIdx.x * 8 + wave;
    if (row >= Nk) return;
    constexpr int NP = W_ / 32;                   // 4 (D=128) or 2 (A=64), contiguous
    const float* xr = X + row * W_ + lane * NP;
    float xv[NP]; float ss = 0.f;
    #pragma unroll
    for (int i = 0; i < NP; ++i) { xv[i] = xr[i]; ss += xv[i] * xv[i]; }
    #pragma unroll
    for (int off = 16; off > 0; off >>= 1) ss += __shfl_xor(ss, off);
    float rr = rsqrtf(ss / (float)W_ + EPSf);
    const float* wp = w + lane * NP;
    float* yr = Y + row * W_ + lane * NP;
    #pragma unroll
    for (int i = 0; i < NP; ++i) yr[i] = xv[i] * rr * wp[i];
}

// ------------------------------- driver --------------------------------
extern "C" void kernel_launch(void* const* d_in, const int* in_sizes, int n_in,
                              void* d_out, int out_size, void* d_ws, size_t ws_size,
                              hipStream_t stream) {
    const float* node  = (const float*)d_in[0];
    const int*   ids   = (const int*)  d_in[1];
    const float* rel   = (const float*)d_in[2];
    const float* Wemb  = (const float*)d_in[3];
    const float* Wq    = (const float*)d_in[4];
    const float* Wk    = (const float*)d_in[5];
    const float* Wv    = (const float*)d_in[6];
    const float* Wedge = (const float*)d_in[7];
    const float* Wskip = (const float*)d_in[8];
    const float* Wout  = (const float*)d_in[9];
    const float* lnw   = (const float*)d_in[10];
    const float* Wact  = (const float*)d_in[11];
    const float* lnf   = (const float*)d_in[12];
    float* out = (float*)d_out;

    char* p = (char*)d_ws;
    auto alloc = [&](size_t bytes) -> char* {
        char* r = p; p += (bytes + 255) & ~(size_t)255; return r;
    };
    float* x0   = (float*)alloc((size_t)Nk * Dk  * 4);
    float* tmp  = (float*)alloc((size_t)Nk * Dk  * 4);
    float* bq   = (float*)alloc((size_t)Nk * HOk * 4);
    float* bk   = (float*)alloc((size_t)Nk * HOk * 4);
    float* bv   = (float*)alloc((size_t)Nk * HOk * 4);
    float* bs   = (float*)alloc((size_t)Nk * HOk * 4);
    float* batn = (float*)alloc((size_t)Nk * HOk * 4);
    float* yb   = (float*)alloc((size_t)Nk * Ak  * 4);
    float* Mbuf = (float*)alloc(2 * HOk * 4);
    int*   deg  = (int*)  alloc((size_t)Nk * 4);
    int*   offs = (int*)  alloc((size_t)(Nk + 1) * 4);
    int*   cur  = (int*)  alloc((size_t)Nk * 4);
    int*   csrc = (int*)  alloc((size_t)Ek * 4);
    float* crx  = (float*)alloc((size_t)Ek * 4);
    float* cry  = (float*)alloc((size_t)Ek * 4);

    hipMemcpyAsync(x0, node, (size_t)Nk * Dk * 4, hipMemcpyDeviceToDevice, stream);

    k_deg_init<<<(Nk + 255) / 256, 256, 0, stream>>>(deg);
    k_count<<<(Nk * Lk + 255) / 256, 256, 0, stream>>>(ids, deg);
    k_scan<<<1, 256, 0, stream>>>(deg, offs, cur);
    k_fill<<<(Nk + Nk * Lk + 255) / 256, 256, 0, stream>>>(ids, rel, cur, csrc, crx, cry);

    for (int r = 0; r < Rr; ++r) {
        const int wOff = r * Dk * HOk;
        k_edgeM<<<2, 512, 0, stream>>>(Wemb + r * 2 * Dk, Wedge + wOff, Mbuf);

        dim3 g1(HOk / 64, Nk / 16);
        k_gemm<Dk, HOk, false><<<g1, 128, 0, stream>>>(x0, Wq    + wOff, nullptr, bq);
        k_gemm<Dk, HOk, false><<<g1, 128, 0, stream>>>(x0, Wk    + wOff, nullptr, bk);
        k_gemm<Dk, HOk, false><<<g1, 128, 0, stream>>>(x0, Wv    + wOff, nullptr, bv);
        k_gemm<Dk, HOk, false><<<g1, 128, 0, stream>>>(x0, Wskip + wOff, nullptr, bs);

        k_attn<<<Nk / 8, 256, 0, stream>>>(bq, bk, bv, bs, Mbuf, offs, csrc, crx, cry, batn);

        dim3 g2(Dk / 64, Nk / 16);
        k_gemm<HOk, Dk, true><<<g2, 128, 0, stream>>>(batn, Wout + r * HOk * Dk, x0, tmp);
        k_rmsnorm<Dk><<<Nk / 8, 256, 0, stream>>>(tmp, lnw + r * Dk, x0);
    }

    dim3 g3(Ak / 64, Nk / 16);
    k_gemm<Dk, Ak, false><<<g3, 128, 0, stream>>>(x0, Wact, nullptr, yb);
    k_rmsnorm<Ak><<<Nk / 8, 256, 0, stream>>>(yb, lnf, out);

    (void)in_sizes; (void)n_in; (void)out_size; (void)ws_size;
}
